// DecoderSoftAttn_82145544503489
// MI455X (gfx1250) — matrix operand order repaired and checked by reading, not compiled, not run on hardware
//
#include <hip/hip_runtime.h>
#include <math.h>

// ---------------- problem constants (from reference) ----------------
#define BB      4
#define LL      512
#define TT      (BB * LL)           // 2048 tokens
#define DM      384                 // d_model
#define DI      768                 // d_inner
#define DSTATE  16
#define DTR     25                  // dt_rank
#define NX      (DTR + 2 * DSTATE)  // 57
#define DCONV   4
#define NMELS   80
#define NLAYER  8
#define WPB     4                   // waves (m-tiles) per block in GEMM
#define NT      4                   // 16-wide N tiles per wave

typedef __bf16 bf16;
typedef __attribute__((ext_vector_type(8)))  __bf16 v8bf;
typedef __attribute__((ext_vector_type(16))) __bf16 v16bf;
typedef __attribute__((ext_vector_type(8)))  float  v8f;

// staged-weight sizes: kchunks * tilesNp * 512 bf16 (tilesNp = tilesN
// rounded up to NT so the GEMM inner loop has no tile guards)
#define WIN_ST_SZ   (3  * 24 * 512)   // w_in   K=80  N=384  (24 tiles)
#define IP_ST_SZ    (12 * 96 * 512)   // inproj K=384 N=1536 (96 tiles)
#define XP_ST_SZ    (24 * 4  * 512)   // xproj  K=768 N=57   (4 tiles)
#define DT_ST_SZ    (1  * 48 * 512)   // dtproj K=25  N=768  (48 tiles)
#define OP_ST_SZ    (24 * 24 * 512)   // outproj K=768 N=384 (24 tiles)
#define WOUT_ST_SZ  (12 * 8  * 512)   // w_out  K=384 N=80   (5 -> 8 tiles)
#define ENCT_ST_SZ  (12 * 32 * 512)   // enc^T  K=384 N=512  (per batch)
#define ENC_ST_SZ   (16 * 24 * 512)   // enc    K=512 N=384  (per batch)

// =====================================================================
// bf16 WMMA GEMM. A: row-major bf16, lda multiple of 32. Bst: fragment-
// major bf16, padded to tilesNp n-tiles (zero-filled). Straight-line
// K-loop: one s_clause of (2 A + 8 B) b128 loads, one wait, then 4
// back-to-back v_wmma. Grid: x = n-tile group, y*WPB+wave = m-tile.
// =====================================================================
__global__ void wmma_gemm(const bf16* __restrict__ A, int lda,
                          const bf16* __restrict__ Bst,
                          float* __restrict__ C, int ldc,
                          bf16* __restrict__ Cbf, int ldcbf,
                          int N, int K,
                          const float* __restrict__ bias,
                          const float* __restrict__ resid, int ldr,
                          int act, int maskK, int tilesNp)
{
    const int lane   = threadIdx.x & 31;
    const int laneHi = lane >> 4;
    const int laneLo = lane & 15;
    const int tm     = blockIdx.y * WPB + (threadIdx.x >> 5);
    const int gn     = blockIdx.x;
    const int kchunks = (K + 31) >> 5;

    // A fragment: lane = m + 16*k[3]; elem = (k&7) + 8*k[4]
    const bf16*  Arow    = A + (size_t)(tm * 16 + laneLo) * lda + (laneHi << 3);
    const bf16*  Bb      = Bst + (size_t)(gn * NT) * 512 + (size_t)lane * 16;
    const size_t bstride = (size_t)tilesNp * 512;

    v8f acc[NT];
#pragma unroll
    for (int t = 0; t < NT; ++t)
#pragma unroll
        for (int r = 0; r < 8; ++r) acc[t][r] = 0.f;

    if (!maskK) {
        for (int kc = 0; kc < kchunks; ++kc) {
            v8bf alo = *(const v8bf*)(Arow + kc * 32);
            v8bf ahi = *(const v8bf*)(Arow + kc * 32 + 16);
            const bf16* bp = Bb + (size_t)kc * bstride;
            v16bf bfrag[NT];
#pragma unroll
            for (int t = 0; t < NT; ++t)
                bfrag[t] = *(const v16bf*)(bp + (size_t)t * 512);
            v16bf a = __builtin_shufflevector(alo, ahi,
                          0,1,2,3,4,5,6,7,8,9,10,11,12,13,14,15);
#pragma unroll
            for (int t = 0; t < NT; ++t)
                acc[t] = __builtin_amdgcn_wmma_f32_16x16x32_bf16(
                             false, a, false, bfrag[t], (short)0, acc[t], false, false);
        }
    } else {                                    // dt-proj: K=25, 1 chunk
        for (int kc = 0; kc < kchunks; ++kc) {
            v8bf alo = *(const v8bf*)(Arow + kc * 32);
            v8bf ahi = *(const v8bf*)(Arow + kc * 32 + 16);
            const bf16* bp = Bb + (size_t)kc * bstride;
            v16bf bfrag[NT];
#pragma unroll
            for (int t = 0; t < NT; ++t)
                bfrag[t] = *(const v16bf*)(bp + (size_t)t * 512);
            int base = kc * 32 + (laneHi << 3);
#pragma unroll
            for (int e = 0; e < 8; ++e) {
                if (base + e >= K)      alo[e] = (bf16)0.f;
                if (base + 16 + e >= K) ahi[e] = (bf16)0.f;
            }
            v16bf a = __builtin_shufflevector(alo, ahi,
                          0,1,2,3,4,5,6,7,8,9,10,11,12,13,14,15);
#pragma unroll
            for (int t = 0; t < NT; ++t)
                acc[t] = __builtin_amdgcn_wmma_f32_16x16x32_bf16(
                             false, a, false, bfrag[t], (short)0, acc[t], false, false);
        }
    }

    // C/D layout: vgpr r -> m = tm*16 + 8*laneHi + r, n = tn*16 + laneLo
#pragma unroll
    for (int t = 0; t < NT; ++t) {
        int  n   = (gn * NT + t) * 16 + laneLo;
        bool nok = (n < N);
        float bv = (bias && nok) ? bias[n] : 0.f;
#pragma unroll
        for (int r = 0; r < 8; ++r) {
            int m = tm * 16 + (laneHi << 3) + r;
            float v = acc[t][r] + bv;
            if (act == 1)      v = v / (1.f + __expf(-v));              // silu
            else if (act == 2) v = (v > 20.f) ? v : log1pf(__expf(v));  // softplus
            if (nok) {
                if (resid) v += resid[(size_t)m * ldr + n];
                C[(size_t)m * ldc + n] = v;
                if (Cbf) Cbf[(size_t)m * ldcbf + n] = (bf16)v;
            }
        }
    }
}

// =====================================================================
// Stage row-major f32 (or its transpose) into WMMA B-fragment layout:
// out[((kc*tilesNp + nt)*32 + lane)*16 + e],
// k = kc*32 + (e&7) + 8*(lane>>4) + 16*(e>>3), n = nt*16 + (lane&15). Zero-pads K/N.
// =====================================================================
__global__ void stage_b_kernel(const float* __restrict__ W, bf16* __restrict__ out,
                               int K, int N, int ldw, int transB,
                               int tilesNp, int total)
{
    int idx = blockIdx.x * blockDim.x + threadIdx.x;
    if (idx >= total) return;
    int e    = idx & 15;
    int lane = (idx >> 4) & 31;
    int tile = idx >> 9;
    int nt   = tile % tilesNp;
    int kc   = tile / tilesNp;
    int k    = kc * 32 + (e & 7) + ((lane >> 4) << 3) + ((e >> 3) << 4);
    int n    = nt * 16 + (lane & 15);
    float v = 0.f;
    if (k < K && n < N)
        v = transB ? W[(size_t)n * ldw + k] : W[(size_t)k * ldw + n];
    out[idx] = (bf16)v;
}

// f32 -> bf16 row-major with K padding (zero fill)
__global__ void cvt_bf16_pad(const float* __restrict__ src, bf16* __restrict__ dst,
                             int Ksrc, int Kdst, int total)
{
    int idx = blockIdx.x * blockDim.x + threadIdx.x;
    if (idx >= total) return;
    int m = idx / Kdst, c = idx % Kdst;
    dst[idx] = (bf16)((c < Ksrc) ? src[(size_t)m * Ksrc + c] : 0.f);
}

// =====================================================================
// RMSNorm: one wave per token, bf16 output (GEMM A operand)
// =====================================================================
__global__ void rmsnorm_kernel(const float* __restrict__ x,
                               const float* __restrict__ w,
                               bf16* __restrict__ out, int T, int D)
{
    int gw   = (blockIdx.x * blockDim.x + threadIdx.x) >> 5;
    int lane = threadIdx.x & 31;
    if (gw >= T) return;
    const float* row = x + (size_t)gw * D;
    float ss = 0.f;
    for (int i = lane; i < D; i += 32) { float v = row[i]; ss += v * v; }
#pragma unroll
    for (int off = 16; off > 0; off >>= 1) ss += __shfl_xor(ss, off, 32);
    float scale = rsqrtf(ss / (float)D + 1e-5f);
    bf16* o = out + (size_t)gw * D;
    for (int i = lane; i < D; i += 32) o[i] = (bf16)(row[i] * scale * w[i]);
}

// =====================================================================
// Depthwise causal conv (k=4) + bias + SiLU; f32 (for scan) + bf16
// (GEMM A). One thread per (batch, channel); window in registers.
// =====================================================================
__global__ void conv_silu_kernel(const float* __restrict__ xr,
                                 const float* __restrict__ cw,
                                 const float* __restrict__ cb,
                                 float* __restrict__ outf,
                                 bf16* __restrict__ outb)
{
    int idx = blockIdx.x * blockDim.x + threadIdx.x;
    if (idx >= BB * DI) return;
    int b = idx / DI, d = idx % DI;
    float w0 = cw[d * DCONV + 0], w1 = cw[d * DCONV + 1];
    float w2 = cw[d * DCONV + 2], w3 = cw[d * DCONV + 3];
    float bias = cb[d];
    float x0 = 0.f, x1 = 0.f, x2 = 0.f;
    const float* src = xr + (size_t)b * LL * (2 * DI) + d;
    size_t o = (size_t)b * LL * DI + d;
    for (int l = 0; l < LL; ++l) {
        float x3 = src[(size_t)l * (2 * DI)];
        float v  = w0 * x0 + w1 * x1 + w2 * x2 + w3 * x3 + bias;
        float s  = v / (1.f + __expf(-v));
        outf[o + (size_t)l * DI] = s;
        outb[o + (size_t)l * DI] = (bf16)s;
        x0 = x1; x1 = x2; x2 = x3;
    }
}

// =====================================================================
// Selective scan: one thread per (batch, channel); 16-state recurrence
// in registers; fused y += u*D and y *= silu(res); bf16 output.
// =====================================================================
__global__ void scan_kernel(const float* __restrict__ delta,
                            const float* __restrict__ xs2,
                            const float* __restrict__ xdbl,
                            const float* __restrict__ xr,
                            const float* __restrict__ A_log,
                            const float* __restrict__ Dp,
                            bf16* __restrict__ ys)
{
    int idx = blockIdx.x * blockDim.x + threadIdx.x;
    if (idx >= BB * DI) return;
    int b = idx / DI, d = idx % DI;

    float Av[DSTATE];
#pragma unroll
    for (int n = 0; n < DSTATE; ++n) Av[n] = -__expf(A_log[d * DSTATE + n]);
    float Dv = Dp[d];
    float h[DSTATE];
#pragma unroll
    for (int n = 0; n < DSTATE; ++n) h[n] = 0.f;

    for (int l = 0; l < LL; ++l) {
        size_t t  = (size_t)(b * LL + l);
        float  dl = delta[t * DI + d];
        float  u  = xs2[t * DI + d];
        const float* bc = xdbl + t * NX;   // [0:25)=dt, [25:41)=B, [41:57)=C
        float y = 0.f;
#pragma unroll
        for (int n = 0; n < DSTATE; ++n) {
            float dA = __expf(dl * Av[n]);
            h[n] = dA * h[n] + dl * bc[DTR + n] * u;
            y += h[n] * bc[DTR + DSTATE + n];
        }
        y += u * Dv;
        float r = xr[t * (2 * DI) + DI + d];
        ys[t * DI + d] = (bf16)(y * (r / (1.f + __expf(-r))));  // y * silu(res)
    }
}

// =====================================================================
// Row softmax (rows of 512): one wave per row, values in registers,
// bf16 probabilities out (GEMM A operand).
// =====================================================================
__global__ void softmax_kernel(const float* __restrict__ s,
                               bf16* __restrict__ p, int rows)
{
    int gw   = (blockIdx.x * blockDim.x + threadIdx.x) >> 5;
    int lane = threadIdx.x & 31;
    if (gw >= rows) return;
    const float* row = s + (size_t)gw * LL;
    float v[16];
#pragma unroll
    for (int i = 0; i < 16; ++i) v[i] = row[lane + i * 32];
    float mx = v[0];
#pragma unroll
    for (int i = 1; i < 16; ++i) mx = fmaxf(mx, v[i]);
#pragma unroll
    for (int off = 16; off > 0; off >>= 1) mx = fmaxf(mx, __shfl_xor(mx, off, 32));
    float sum = 0.f;
#pragma unroll
    for (int i = 0; i < 16; ++i) { v[i] = __expf(v[i] - mx); sum += v[i]; }
#pragma unroll
    for (int off = 16; off > 0; off >>= 1) sum += __shfl_xor(sum, off, 32);
    float inv = 1.f / sum;
    bf16* o = p + (size_t)gw * LL;
#pragma unroll
    for (int i = 0; i < 16; ++i) o[lane + i * 32] = (bf16)(v[i] * inv);
}

// =====================================================================
extern "C" void kernel_launch(void* const* d_in, const int* in_sizes, int n_in,
                              void* d_out, int out_size, void* d_ws, size_t ws_size,
                              hipStream_t stream)
{
    const float* x_in  = (const float*)d_in[0];
    const float* enc   = (const float*)d_in[1];
    const float* w_in  = (const float*)d_in[2];
    const float* b_in  = (const float*)d_in[3];
    const float* ln_w  = (const float*)d_in[4];   // [8][384]
    const float* ip_w  = (const float*)d_in[5];   // [8][384][1536]
    const float* cv_w  = (const float*)d_in[6];   // [8][768][4]
    const float* cv_b  = (const float*)d_in[7];   // [8][768]
    const float* xp_w  = (const float*)d_in[8];   // [8][768][57]
    const float* dt_w  = (const float*)d_in[9];   // [8][25][768]
    const float* dt_b  = (const float*)d_in[10];  // [8][768]
    const float* A_log = (const float*)d_in[11];  // [8][768][16]
    const float* Dp    = (const float*)d_in[12];  // [8][768]
    const float* op_w  = (const float*)d_in[13];  // [8][768][384]
    const float* nf_w  = (const float*)d_in[14];  // [384]
    const float* wout  = (const float*)d_in[15];  // [384][80]
    float* out = (float*)d_out;

    // ---- carve workspace (256B-aligned slices) ----
    char* p = (char*)d_ws;
    auto alloc_f = [&](size_t n) { float* r = (float*)p; p += ((n * 4 + 255) / 256) * 256; return r; };
    auto alloc_b = [&](size_t n) { bf16*  r = (bf16*)p;  p += ((n * 2 + 255) / 256) * 256; return r; };

    float* h      = alloc_f((size_t)TT * DM);
    float* xr     = alloc_f((size_t)TT * 2 * DI);
    float* xs2    = alloc_f((size_t)TT * DI);
    float* xdbl   = alloc_f((size_t)TT * NX);
    float* dlt    = alloc_f((size_t)TT * DI);
    float* scores = alloc_f((size_t)BB * LL * LL);

    bf16* x_bf    = alloc_b((size_t)TT * 96);   // K=80 padded to 96
    bf16* hn_bf   = alloc_b((size_t)TT * DM);
    bf16* xs2_bf  = alloc_b((size_t)TT * DI);
    bf16* xdbl_bf = alloc_b((size_t)TT * 64);   // NX=57 padded stride 64
    bf16* ys_bf   = alloc_b((size_t)TT * DI);
    bf16* h_bf    = alloc_b((size_t)TT * DM);
    bf16* pr_bf   = alloc_b((size_t)BB * LL * LL);

    bf16* win_st  = alloc_b(WIN_ST_SZ);
    bf16* ip_st   = alloc_b((size_t)NLAYER * IP_ST_SZ);
    bf16* xp_st   = alloc_b((size_t)NLAYER * XP_ST_SZ);
    bf16* dt_st   = alloc_b((size_t)NLAYER * DT_ST_SZ);
    bf16* op_st   = alloc_b((size_t)NLAYER * OP_ST_SZ);
    bf16* wout_st = alloc_b(WOUT_ST_SZ);
    bf16* encT_st = alloc_b((size_t)BB * ENCT_ST_SZ);
    bf16* enc_st  = alloc_b((size_t)BB * ENC_ST_SZ);

    auto tilesNpOf = [](int N) { int t = (N + 15) >> 4; return ((t + NT - 1) / NT) * NT; };

    auto stage = [&](const float* W, bf16* dst, int K, int N, int ldw, int trans) {
        int tnp   = tilesNpOf(N);
        int total = ((K + 31) >> 5) * tnp * 512;
        stage_b_kernel<<<(total + 255) / 256, 256, 0, stream>>>(W, dst, K, N, ldw, trans, tnp, total);
    };
    auto gemm = [&](const bf16* A, int lda, const bf16* Bst, float* C, int ldc,
                    bf16* Cbf, int ldcbf, int M, int N, int K,
                    const float* bias, const float* resid, int ldr, int act, int maskK) {
        int tnp = tilesNpOf(N);
        dim3 grid(tnp / NT, (M / 16) / WPB);    // all M here are multiples of 64
        wmma_gemm<<<grid, WPB * 32, 0, stream>>>(A, lda, Bst, C, ldc, Cbf, ldcbf,
                                                 N, K, bias, resid, ldr, act, maskK, tnp);
    };

    // ---- stage all constant operands to bf16 (fragment layout) ----
    stage(w_in, win_st, NMELS, DM, DM, 0);
    for (int i = 0; i < NLAYER; ++i) {
        stage(ip_w + (size_t)i * DM * 2 * DI, ip_st + (size_t)i * IP_ST_SZ, DM, 2 * DI, 2 * DI, 0);
        stage(xp_w + (size_t)i * DI * NX,     xp_st + (size_t)i * XP_ST_SZ, DI, NX, NX, 0);
        stage(dt_w + (size_t)i * DTR * DI,    dt_st + (size_t)i * DT_ST_SZ, DTR, DI, DI, 0);
        stage(op_w + (size_t)i * DI * DM,     op_st + (size_t)i * OP_ST_SZ, DI, DM, DM, 0);
    }
    stage(wout, wout_st, DM, NMELS, NMELS, 0);
    for (int b = 0; b < BB; ++b) {
        stage(enc + (size_t)b * LL * DM, encT_st + (size_t)b * ENCT_ST_SZ, DM, LL, DM, 1); // enc^T
        stage(enc + (size_t)b * LL * DM, enc_st  + (size_t)b * ENC_ST_SZ,  LL, DM, DM, 0); // enc
    }
    {   // x -> bf16, K padded 80 -> 96 (zero fill)
        int total = TT * 96;
        cvt_bf16_pad<<<(total + 255) / 256, 256, 0, stream>>>(x_in, x_bf, NMELS, 96, total);
    }

    // h = x @ w_in + b_in
    gemm(x_bf, 96, win_st, h, DM, nullptr, 0, TT, DM, NMELS, b_in, nullptr, 0, 0, 0);

    auto mamba_layer = [&](int i) {
        rmsnorm_kernel<<<(TT * 32 + 255) / 256, 256, 0, stream>>>(h, ln_w + (size_t)i * DM, hn_bf, TT, DM);
        // xr = hn @ inproj
        gemm(hn_bf, DM, ip_st + (size_t)i * IP_ST_SZ, xr, 2 * DI, nullptr, 0,
             TT, 2 * DI, DM, nullptr, nullptr, 0, 0, 0);
        conv_silu_kernel<<<(BB * DI + 255) / 256, 256, 0, stream>>>(
            xr, cv_w + (size_t)i * DI * DCONV, cv_b + (size_t)i * DI, xs2, xs2_bf);
        // xdbl = xs2 @ xproj (f32 for scan + bf16 copy for dt gemm)
        gemm(xs2_bf, DI, xp_st + (size_t)i * XP_ST_SZ, xdbl, NX, xdbl_bf, 64,
             TT, NX, DI, nullptr, nullptr, 0, 0, 0);
        // delta = softplus(dt @ dtproj + db); A = xdbl_bf[:, :25] (masked K)
        gemm(xdbl_bf, 64, dt_st + (size_t)i * DT_ST_SZ, dlt, DI, nullptr, 0,
             TT, DI, DTR, dt_b + (size_t)i * DI, nullptr, 0, /*softplus*/2, /*maskK*/1);
        scan_kernel<<<(BB * DI + 255) / 256, 256, 0, stream>>>(
            dlt, xs2, xdbl, xr, A_log + (size_t)i * DI * DSTATE, Dp + (size_t)i * DI, ys_bf);
        // h = h + ys @ outproj  (+ bf16 copy of h for attention A operand)
        gemm(ys_bf, DI, op_st + (size_t)i * OP_ST_SZ, h, DM, h_bf, DM,
             TT, DM, DI, nullptr, /*resid*/h, DM, 0, 0);
    };

    for (int i = 0; i < NLAYER / 2; ++i) mamba_layer(i);

    // cross attention: scores = h @ enc^T ; softmax ; h = P @ enc
    for (int b = 0; b < BB; ++b)
        gemm(h_bf + (size_t)b * LL * DM, DM, encT_st + (size_t)b * ENCT_ST_SZ,
             scores + (size_t)b * LL * LL, LL, nullptr, 0,
             LL, LL, DM, nullptr, nullptr, 0, 0, 0);
    softmax_kernel<<<(BB * LL * 32 + 255) / 256, 256, 0, stream>>>(scores, pr_bf, BB * LL);
    for (int b = 0; b < BB; ++b)
        gemm(pr_bf + (size_t)b * LL * LL, LL, enc_st + (size_t)b * ENC_ST_SZ,
             h + (size_t)b * LL * DM, DM, nullptr, 0,
             LL, DM, LL, nullptr, nullptr, 0, 0, 0);

    for (int i = NLAYER / 2; i < NLAYER; ++i) mamba_layer(i);

    // final norm + output projection
    rmsnorm_kernel<<<(TT * 32 + 255) / 256, 256, 0, stream>>>(h, nf_w, hn_bf, TT, DM);
    gemm(hn_bf, DM, wout_st, out, NMELS, nullptr, 0,
         TT, NMELS, DM, nullptr, nullptr, 0, 0, 0);
}
